// FloodGNN_56899726737415
// MI455X (gfx1250) — matrix operand order repaired and checked by reading, not compiled, not run on hardware
//
#include <hip/hip_runtime.h>
#include <hip/hip_bf16.h>

typedef __attribute__((ext_vector_type(16))) _Float16 v16h;
typedef __attribute__((ext_vector_type(8)))  float    v8f;

#define HID 64

__device__ __forceinline__ int imin(int a, int b){ return a < b ? a : b; }

// ---------------------------------------------------------------- elementwise
__global__ void zero_f(float* __restrict__ p, int n){
  int i = blockIdx.x * blockDim.x + threadIdx.x;
  if (i < n) p[i] = 0.0f;
}

__global__ void degree_f(const int* __restrict__ dst, float* __restrict__ deg, int e){
  int i = blockIdx.x * blockDim.x + threadIdx.x;
  if (i < e) atomicAdd(&deg[dst[i]], 1.0f);
}

__global__ void invdeg_f(float* __restrict__ deg, int n){
  int i = blockIdx.x * blockDim.x + threadIdx.x;
  if (i < n){ float d = deg[i]; deg[i] = 1.0f / (d > 1.0f ? d : 1.0f); }
}

// gather feat[src[e]] (64 floats, L2-hot) and atomically accumulate into acc[dst[e]]
// 16 threads per edge, float4 per thread -> 4 global_atomic_add_f32 each
__global__ void scatter_add64(const float* __restrict__ feat, const int* __restrict__ src,
                              const int* __restrict__ dst, float* __restrict__ acc, int e){
  int t = blockIdx.x * blockDim.x + threadIdx.x;
  if (t >= e * 16) return;
  int ed = t >> 4, q = t & 15;
  int s = src[ed], d = dst[ed];
  float4 v = ((const float4*)(feat + (size_t)s * HID))[q];
  float* o = acc + (size_t)d * HID + q * 4;
  atomicAdd(o + 0, v.x); atomicAdd(o + 1, v.y);
  atomicAdd(o + 2, v.z); atomicAdd(o + 3, v.w);
}

// acc = relu(acc * inv_deg[node] + self)
__global__ void combine_relu(float* __restrict__ acc, const float* __restrict__ self,
                             const float* __restrict__ inv, int n64){
  int t = blockIdx.x * blockDim.x + threadIdx.x;
  if (t >= n64) return;
  float v = acc[t] * inv[t >> 6] + self[t];
  acc[t] = v > 0.0f ? v : 0.0f;
}

// out[i] = h[i,:] . Wc + bc
__global__ void classify_k(const float* __restrict__ h, const float* __restrict__ Wc,
                           const float* __restrict__ bc, float* __restrict__ out, int n){
  __shared__ float wc[HID];
  if (threadIdx.x < HID) wc[threadIdx.x] = Wc[threadIdx.x];
  __syncthreads();
  int i = blockIdx.x * blockDim.x + threadIdx.x;
  if (i >= n) return;
  const float4* hp = (const float4*)(h + (size_t)i * HID);
  float s = bc[0];
#pragma unroll
  for (int j = 0; j < 16; ++j){
    float4 v = hp[j];
    s += v.x * wc[4*j+0] + v.y * wc[4*j+1] + v.z * wc[4*j+2] + v.w * wc[4*j+3];
  }
  out[i] = s;
}

// ---------------------------------------------------------------- WMMA GEMM
// Yl = X @ Wl ; Yr = X @ Wr + bias.   X:[nrows,K] f32, W:[K,64] f32, Y:[nrows,64] f32
// 256 threads = 8 waves; wave w owns rows [blk*128 + 16w, +16), all 64 cols (4 tiles).
// Weights staged in LDS as f16 pre-swizzled into the B-fragment lane layout so each
// lane reads its v16h B fragment as one contiguous 32-byte LDS access.
template<int K>
__global__ __launch_bounds__(256)
void gemm_dual_wmma(const float* __restrict__ X,
                    const float* __restrict__ Wl, const float* __restrict__ Wr,
                    const float* __restrict__ bias,
                    float* __restrict__ Yl, float* __restrict__ Yr, int nrows){
  __shared__ __align__(32) _Float16 sW[2 * K * HID];   // [mat][kh][col][j:16]
  // swizzled fill: element j of fragment (kh, col) is W[(kh/2)*32 + (j&8?16:0) + (kh&1)*8 + (j&7)][col]
  for (int idx = threadIdx.x; idx < K * HID; idx += 256){
    int j   = idx & 15;
    int col = (idx >> 4) & 63;
    int kh  = idx >> 10;
    int k   = (kh >> 1) * 32 + ((j & 8) ? 16 : 0) + (kh & 1) * 8 + (j & 7);
    sW[idx]           = (_Float16)Wl[k * HID + col];
    sW[K*HID + idx]   = (_Float16)Wr[k * HID + col];
  }
  __syncthreads();

  int wave = threadIdx.x >> 5;
  int lane = threadIdx.x & 31;
  int hfl  = lane >> 4;          // lane half: selects K sub-range per ISA layout
  int lr   = lane & 15;
  int row0 = blockIdx.x * 128 + wave * 16;
  if (row0 >= nrows) return;     // wave-uniform; WMMA waves keep EXEC all-ones

  int arow = imin(row0 + lr, nrows - 1);
  const float* xrow = X + (size_t)arow * K;

  v8f accl[4] = {}, accr[4] = {};
#pragma unroll
  for (int kb = 0; kb < K; kb += 32){
    if (kb + 32 < K) __builtin_prefetch(xrow + kb + 32, 0, 1);
    // A fragment (16-bit A 16x32 layout): a[j] = X[row, kb + (j&8?16:0) + hfl*8 + (j&7)]
    v16h a;
#pragma unroll
    for (int j = 0; j < 16; j += 2){
      int k = kb + ((j & 8) ? 16 : 0) + hfl * 8 + (j & 7);
      float2 xv = *(const float2*)(xrow + k);
      a[j]   = (_Float16)xv.x;
      a[j+1] = (_Float16)xv.y;
    }
    int kh = (kb >> 5) * 2 + hfl;
#pragma unroll
    for (int t = 0; t < 4; ++t){
      int col = t * 16 + lr;
      v16h bl = *(const v16h*)(sW + (size_t)(kh * 64 + col) * 16);
      v16h br = *(const v16h*)(sW + (size_t)K*HID + (size_t)(kh * 64 + col) * 16);
      accl[t] = __builtin_amdgcn_wmma_f32_16x16x32_f16(false, a, false, bl,
                                                       (short)0, accl[t], false, false);
      accr[t] = __builtin_amdgcn_wmma_f32_16x16x32_f16(false, a, false, br,
                                                       (short)0, accr[t], false, false);
    }
  }
  // C/D layout: VGPR r of lane -> (M = r + 8*hfl, N = lr)
#pragma unroll
  for (int t = 0; t < 4; ++t){
    int col = t * 16 + lr;
    float bv = bias[col];
#pragma unroll
    for (int r = 0; r < 8; ++r){
      int m = row0 + r + 8 * hfl;
      if (m < nrows){
        Yl[(size_t)m * HID + col] = accl[t][r];
        Yr[(size_t)m * HID + col] = accr[t][r] + bv;
      }
    }
  }
}

// ---------------------------------------------------------------- launch
extern "C" void kernel_launch(void* const* d_in, const int* in_sizes, int n_in,
                              void* d_out, int out_size, void* d_ws, size_t ws_size,
                              hipStream_t stream){
  (void)n_in; (void)out_size; (void)ws_size;
  const float* x   = (const float*)d_in[0];
  const int* eidx  = (const int*)d_in[1];
  const float* W1l = (const float*)d_in[2];
  const float* W1r = (const float*)d_in[3];
  const float* b1  = (const float*)d_in[4];
  const float* W2l = (const float*)d_in[5];
  const float* W2r = (const float*)d_in[6];
  const float* b2  = (const float*)d_in[7];
  const float* Wc  = (const float*)d_in[8];
  const float* bc  = (const float*)d_in[9];
  float* out = (float*)d_out;

  int N = in_sizes[0] / 128;
  int E = in_sizes[1] / 2;
  const int* src = eidx;       // edge_index[0]
  const int* dst = eidx + E;   // edge_index[1]

  float* ws  = (float*)d_ws;
  float* inv = ws;                        // [N]     degree -> inv_degree
  float* B0  = ws + N;                    // [N,64]  x@W1_l   | h1@W2_l
  float* B1  = B0 + (size_t)N * 64;       // [N,64]  x@W1_r+b | h1@W2_r+b
  float* B2  = B1 + (size_t)N * 64;       // [N,64]  agg/h1   | agg2/h2

  int n64 = N * 64;
  dim3 blk(256);
  auto g = [](int n){ return dim3((unsigned)((n + 255) / 256)); };
  dim3 ggrid((unsigned)((N + 127) / 128));

  // degree / inverse degree
  zero_f   <<<g(N),   blk, 0, stream>>>(inv, N);
  zero_f   <<<g(n64), blk, 0, stream>>>(B2, n64);
  degree_f <<<g(E),   blk, 0, stream>>>(dst, inv, E);
  invdeg_f <<<g(N),   blk, 0, stream>>>(inv, N);

  // layer 1: project FIRST (linearity of segment_sum), then aggregate 64-dim rows
  gemm_dual_wmma<128><<<ggrid, blk, 0, stream>>>(x, W1l, W1r, b1, B0, B1, N);
  scatter_add64<<<g(E * 16), blk, 0, stream>>>(B0, src, dst, B2, E);
  combine_relu <<<g(n64),    blk, 0, stream>>>(B2, B1, inv, n64);   // B2 = h1

  // layer 2
  gemm_dual_wmma<64><<<ggrid, blk, 0, stream>>>(B2, W2l, W2r, b2, B0, B1, N);
  zero_f       <<<g(n64),    blk, 0, stream>>>(B2, n64);
  scatter_add64<<<g(E * 16), blk, 0, stream>>>(B0, src, dst, B2, E);
  combine_relu <<<g(n64),    blk, 0, stream>>>(B2, B1, inv, n64);   // B2 = h2

  // classifier
  classify_k<<<g(N), blk, 0, stream>>>(B2, Wc, bc, out, N);
}
